// MultiHeadAttention_22428319220498
// MI455X (gfx1250) — matrix operand order repaired and checked by reading, not compiled, not run on hardware
//
#include <hip/hip_runtime.h>
#include <hip/hip_bf16.h>

typedef __bf16 bf16;
typedef __attribute__((ext_vector_type(16))) __bf16 v16bf;
typedef __attribute__((ext_vector_type(8)))  __bf16 v8bf;
typedef __attribute__((ext_vector_type(8)))  float  v8f;

#define WMMA_BF16(A, B, C) \
    __builtin_amdgcn_wmma_f32_16x16x32_bf16(false, (A), false, (B), (short)0, (C), false, false)

__device__ __forceinline__ v16bf join8(v8bf lo, v8bf hi) {
    union { v16bf v; v8bf h[2]; } u;
    u.h[0] = lo; u.h[1] = hi;
    return u.v;
}

__device__ __forceinline__ v8f zero8() {
    v8f z = {0.f, 0.f, 0.f, 0.f, 0.f, 0.f, 0.f, 0.f};
    return z;
}

// ---------------------------------------------------------------------------
// fp32 -> bf16 conversion, 8 elements per thread (b128-pair load, b128 store)
// n must be a multiple of 8 (all tensors here are multiples of 1024).
// ---------------------------------------------------------------------------
__global__ __launch_bounds__(256)
void cvt_f32_bf16x8(const float* __restrict__ in, bf16* __restrict__ out, int n) {
    const int i = (blockIdx.x * blockDim.x + threadIdx.x) * 8;
    if (i < n) {
        v8f x = *(const v8f*)(in + i);
        v8bf y;
#pragma unroll
        for (int j = 0; j < 8; ++j) y[j] = (bf16)x[j];
        *(v8bf*)(out + i) = y;
    }
}

// ---------------------------------------------------------------------------
// Projection GEMM: Y = (X @ W^T + bias) * scale
//   X: (M=4096, K=1024) bf16 row-major, W: (Nout=1024, K=1024) bf16 row-major
//   Each wave computes a 64x64 tile (16 WMMA accumulators). 8 waves/block.
//   mode 0: store (B,H,N,D)   Y[((b*16+h)*2048+n)*64+d]     (Q, K)
//   mode 1: store (B,H,D,N)   Y[((b*16+h)*64+d)*2048+n]     (V transposed,
//           vectorized b128 store: 8 consecutive n per lane)
// ---------------------------------------------------------------------------
__global__ __launch_bounds__(256)
void proj_gemm_bf16(const bf16* __restrict__ X, const bf16* __restrict__ W,
                    const float* __restrict__ bias, bf16* __restrict__ Y,
                    int K, int tilesN, float scale, int mode)
{
    const int lane = threadIdx.x & 31;
    const int widx = threadIdx.x >> 5;
    const int tile = blockIdx.x * 8 + widx;
    const int m0 = (tile / tilesN) * 64;
    const int n0 = (tile % tilesN) * 64;
    const int l15 = lane & 15;
    const int hi  = lane >> 4;

    v8f acc[4][4];
#pragma unroll
    for (int i = 0; i < 4; ++i)
#pragma unroll
        for (int j = 0; j < 4; ++j) acc[i][j] = zero8();

    for (int k0 = 0; k0 < K; k0 += 32) {
        v16bf a[4], b[4];
#pragma unroll
        for (int mt = 0; mt < 4; ++mt) {
            const bf16* p = X + (size_t)(m0 + mt * 16 + l15) * K + k0 + 8 * hi;
            a[mt] = join8(*(const v8bf*)p, *(const v8bf*)(p + 16));
        }
#pragma unroll
        for (int nt = 0; nt < 4; ++nt) {
            const bf16* p = W + (size_t)(n0 + nt * 16 + l15) * K + k0 + 8 * hi;
            b[nt] = join8(*(const v8bf*)p, *(const v8bf*)(p + 16));
        }
#pragma unroll
        for (int mt = 0; mt < 4; ++mt)
#pragma unroll
            for (int nt = 0; nt < 4; ++nt)
                acc[mt][nt] = WMMA_BF16(a[mt], b[nt], acc[mt][nt]);
    }

#pragma unroll
    for (int nt = 0; nt < 4; ++nt) {
        const int f = n0 + nt * 16 + l15;
        const int h = f >> 6, d = f & 63;
        const float bv = bias[f];
#pragma unroll
        for (int mt = 0; mt < 4; ++mt) {
            if (mode == 0) {
#pragma unroll
                for (int j = 0; j < 8; ++j) {
                    const int m = m0 + mt * 16 + j + 8 * hi;
                    const int bb = m >> 11, n = m & 2047;
                    const float v = (acc[mt][nt][j] + bv) * scale;
                    Y[((size_t)(bb * 16 + h) * 2048 + n) * 64 + d] = (bf16)v;
                }
            } else {
                // (B,H,D,N): per lane the 8 rows are 8 consecutive n values
                const int mbase = m0 + mt * 16 + 8 * hi;
                const int bb = mbase >> 11, nb = mbase & 2047;
                v8bf pk;
#pragma unroll
                for (int j = 0; j < 8; ++j)
                    pk[j] = (bf16)((acc[mt][nt][j] + bv) * scale);
                *(v8bf*)(Y + ((size_t)(bb * 16 + h) * 64 + d) * 2048 + nb) = pk;
            }
        }
    }
}

// ---------------------------------------------------------------------------
// Flash attention (causal, online softmax). One wave per (bh, 32-row Q block):
// two 16-row q-tiles share every K A-fragment and V B-fragment, giving
// 16 WMMAs per 16 b128 loads per 32-key block.
//   Q,K: (B,H,N,D) bf16 (Q pre-scaled by 1/sqrt(D)); Vt: (B,H,D,N) bf16.
//   S^T = K_blk(32xD) . Q^T(Dx16): the S^T C-layout converts to the P
//   A-fragment with no cross-lane movement; O += P(16x32) . V(32x64).
//   Out: (B,N,E) bf16.
// ---------------------------------------------------------------------------
__global__ __launch_bounds__(256)
void flash_attn(const bf16* __restrict__ Q, const bf16* __restrict__ Kk,
                const bf16* __restrict__ Vt, bf16* __restrict__ Out)
{
    const int lane = threadIdx.x & 31;
    const int widx = threadIdx.x >> 5;
    const int wg = blockIdx.x * 8 + widx;   // 2048 waves total
    const int bh = wg >> 6;                 // 0..31
    const int qt = wg & 63;
    const int q0 = qt << 5;                 // 32 q rows per wave
    const int l15 = lane & 15;
    const int hi  = lane >> 4;

    const bf16* Qp = Q  + (size_t)bh * 2048 * 64;
    const bf16* Kp = Kk + (size_t)bh * 2048 * 64;
    const bf16* Vp = Vt + (size_t)bh * 64 * 2048;

    // Q^T B-fragments, resident across all key blocks: B[k=d][n=qrow]
    v16bf qb[2][2];
#pragma unroll
    for (int u = 0; u < 2; ++u)
#pragma unroll
        for (int ks = 0; ks < 2; ++ks) {
            const bf16* p = Qp + (size_t)(q0 + u * 16 + l15) * 64 + ks * 32 + 8 * hi;
            qb[u][ks] = join8(*(const v8bf*)p, *(const v8bf*)(p + 16));
        }

    float m_i[2] = {-3.0e38f, -3.0e38f};   // running row max (qrow = lane&15)
    float l_i[2] = {0.f, 0.f};             // running row sum
    v8f o[2][4];
#pragma unroll
    for (int u = 0; u < 2; ++u)
#pragma unroll
        for (int dt = 0; dt < 4; ++dt) o[u][dt] = zero8();

    const int nkb = (q0 + 63) >> 5;        // 32-key blocks covering the diagonal

    for (int kb = 0; kb < nkb; ++kb) {
        const int k0 = kb * 32;

        // S^T = K_block . Q^T for both q-tiles; K fragments shared
        v8f s[2][2];
#pragma unroll
        for (int u = 0; u < 2; ++u) { s[u][0] = zero8(); s[u][1] = zero8(); }
#pragma unroll
        for (int ks = 0; ks < 2; ++ks) {
            v16bf a[2];
#pragma unroll
            for (int mt = 0; mt < 2; ++mt) {
                const bf16* p = Kp + (size_t)(k0 + mt * 16 + l15) * 64 + ks * 32 + 8 * hi;
                a[mt] = join8(*(const v8bf*)p, *(const v8bf*)(p + 16));
            }
#pragma unroll
            for (int u = 0; u < 2; ++u)
#pragma unroll
                for (int mt = 0; mt < 2; ++mt)
                    s[u][mt] = WMMA_BF16(a[mt], qb[u][ks], s[u][mt]);
        }

        // Online softmax per q-tile
        v16bf pa[2];
#pragma unroll
        for (int u = 0; u < 2; ++u) {
            const int q_abs = q0 + u * 16 + l15;
            float mx = -3.0e38f;
#pragma unroll
            for (int t = 0; t < 2; ++t) {
#pragma unroll
                for (int j = 0; j < 8; ++j) {
                    const int key = k0 + t * 16 + j + 8 * hi;
                    float sv = s[u][t][j];
                    if (key > q_abs) sv = -3.0e38f;
                    s[u][t][j] = sv;
                    mx = fmaxf(mx, sv);
                }
            }
            mx = fmaxf(mx, __shfl_xor(mx, 16, 32));

            const float m_new = fmaxf(m_i[u], mx);
            const float alpha = __expf(m_i[u] - m_new);

            float rs = 0.f;
#pragma unroll
            for (int t = 0; t < 2; ++t) {
#pragma unroll
                for (int j = 0; j < 8; ++j) {
                    const float pv = __expf(s[u][t][j] - m_new);
                    rs += pv;
                    pa[u][t * 8 + j] = (bf16)pv;
                }
            }
            rs += __shfl_xor(rs, 16, 32);
            l_i[u] = l_i[u] * alpha + rs;
            m_i[u] = m_new;

            // Rescale O: alpha indexed by qrow=lane -> broadcast to row=j+8*hi
#pragma unroll
            for (int j = 0; j < 8; ++j) {
                const float aj = __shfl(alpha, j + 8 * hi, 32);
#pragma unroll
                for (int dt = 0; dt < 4; ++dt) o[u][dt][j] *= aj;
            }
        }

        // O += P . V ; V fragments shared by both q-tiles
#pragma unroll
        for (int dt = 0; dt < 4; ++dt) {
            const bf16* p = Vp + (size_t)(dt * 16 + l15) * 2048 + k0 + 8 * hi;
            v16bf vb = join8(*(const v8bf*)p, *(const v8bf*)(p + 16));
#pragma unroll
            for (int u = 0; u < 2; ++u)
                o[u][dt] = WMMA_BF16(pa[u], vb, o[u][dt]);
        }
    }

    // Normalize by row sum and store (B,N,E) bf16
    const int b = bh >> 4, h = bh & 15;
#pragma unroll
    for (int u = 0; u < 2; ++u) {
        const float linv = 1.0f / l_i[u];
#pragma unroll
        for (int j = 0; j < 8; ++j) {
            const float lj = __shfl(linv, j + 8 * hi, 32);
            const int q = q0 + u * 16 + j + 8 * hi;
            const size_t base = ((size_t)b * 2048 + q) * 1024 + h * 64;
#pragma unroll
            for (int dt = 0; dt < 4; ++dt)
                Out[base + dt * 16 + l15] = (bf16)(o[u][dt][j] * lj);
        }
    }
}

// ---------------------------------------------------------------------------
// Output projection: out = X @ Wo^T + bo, fp32 row-major result
// ---------------------------------------------------------------------------
__global__ __launch_bounds__(256)
void out_gemm_f32(const bf16* __restrict__ X, const bf16* __restrict__ W,
                  const float* __restrict__ bias, float* __restrict__ Y,
                  int K, int Nout, int tilesN)
{
    const int lane = threadIdx.x & 31;
    const int widx = threadIdx.x >> 5;
    const int tile = blockIdx.x * 8 + widx;
    const int m0 = (tile / tilesN) * 64;
    const int n0 = (tile % tilesN) * 64;
    const int l15 = lane & 15;
    const int hi  = lane >> 4;

    v8f acc[4][4];
#pragma unroll
    for (int i = 0; i < 4; ++i)
#pragma unroll
        for (int j = 0; j < 4; ++j) acc[i][j] = zero8();

    for (int k0 = 0; k0 < K; k0 += 32) {
        v16bf a[4], b[4];
#pragma unroll
        for (int mt = 0; mt < 4; ++mt) {
            const bf16* p = X + (size_t)(m0 + mt * 16 + l15) * K + k0 + 8 * hi;
            a[mt] = join8(*(const v8bf*)p, *(const v8bf*)(p + 16));
        }
#pragma unroll
        for (int nt = 0; nt < 4; ++nt) {
            const bf16* p = W + (size_t)(n0 + nt * 16 + l15) * K + k0 + 8 * hi;
            b[nt] = join8(*(const v8bf*)p, *(const v8bf*)(p + 16));
        }
#pragma unroll
        for (int mt = 0; mt < 4; ++mt)
#pragma unroll
            for (int nt = 0; nt < 4; ++nt)
                acc[mt][nt] = WMMA_BF16(a[mt], b[nt], acc[mt][nt]);
    }

#pragma unroll
    for (int nt = 0; nt < 4; ++nt) {
        const int f = n0 + nt * 16 + l15;
        const float bv = bias[f];
#pragma unroll
        for (int mt = 0; mt < 4; ++mt) {
#pragma unroll
            for (int j = 0; j < 8; ++j) {
                const int m = m0 + mt * 16 + j + 8 * hi;
                Y[(size_t)m * Nout + f] = acc[mt][nt][j] + bv;
            }
        }
    }
}

// ---------------------------------------------------------------------------
// Host-side orchestration
// ---------------------------------------------------------------------------
extern "C" void kernel_launch(void* const* d_in, const int* in_sizes, int n_in,
                              void* d_out, int out_size, void* d_ws, size_t ws_size,
                              hipStream_t stream) {
    (void)in_sizes; (void)n_in; (void)out_size; (void)ws_size;

    const float* hidden = (const float*)d_in[0];
    const float* w_q = (const float*)d_in[1];
    const float* b_q = (const float*)d_in[2];
    const float* w_k = (const float*)d_in[3];
    const float* b_k = (const float*)d_in[4];
    const float* w_v = (const float*)d_in[5];
    const float* b_v = (const float*)d_in[6];
    const float* w_o = (const float*)d_in[7];
    const float* b_o = (const float*)d_in[8];

    // Workspace layout (bf16 buffers), sizes in bytes
    char* ws = (char*)d_ws;
    bf16* Xb  = (bf16*)(ws);                          //  8 MiB: (4096,1024)
    bf16* Wqb = (bf16*)(ws + (8u  << 20));            //  2 MiB
    bf16* Wkb = (bf16*)(ws + (10u << 20));            //  2 MiB
    bf16* Wvb = (bf16*)(ws + (12u << 20));            //  2 MiB
    bf16* Wob = (bf16*)(ws + (14u << 20));            //  2 MiB
    bf16* Qh  = (bf16*)(ws + (16u << 20));            //  8 MiB: (B,H,N,D)
    bf16* Kh  = (bf16*)(ws + (24u << 20));            //  8 MiB: (B,H,N,D)
    bf16* Vth = (bf16*)(ws + (32u << 20));            //  8 MiB: (B,H,D,N)
    bf16* Ab  = (bf16*)(ws + (40u << 20));            //  8 MiB: (B,N,E)

    const int NX = 4096 * 1024;   // activations
    const int NW = 1024 * 1024;   // each weight

    cvt_f32_bf16x8<<<NX / 8 / 256, 256, 0, stream>>>(hidden, Xb, NX);
    cvt_f32_bf16x8<<<NW / 8 / 256, 256, 0, stream>>>(w_q, Wqb, NW);
    cvt_f32_bf16x8<<<NW / 8 / 256, 256, 0, stream>>>(w_k, Wkb, NW);
    cvt_f32_bf16x8<<<NW / 8 / 256, 256, 0, stream>>>(w_v, Wvb, NW);
    cvt_f32_bf16x8<<<NW / 8 / 256, 256, 0, stream>>>(w_o, Wob, NW);

    // QKV projections: 64 row-tiles x 16 col-tiles = 1024 wave-tiles -> 128 blocks
    const float qscale = 0.125f;  // 1/sqrt(D), folded into Q (incl. bias)
    proj_gemm_bf16<<<128, 256, 0, stream>>>(Xb, Wqb, b_q, Qh,  1024, 16, qscale, 0);
    proj_gemm_bf16<<<128, 256, 0, stream>>>(Xb, Wkb, b_k, Kh,  1024, 16, 1.0f,   0);
    proj_gemm_bf16<<<128, 256, 0, stream>>>(Xb, Wvb, b_v, Vth, 1024, 16, 1.0f,   1);

    // Attention: 32 (b,h) x 64 q-blocks of 32 rows = 2048 waves -> 256 blocks
    flash_attn<<<256, 256, 0, stream>>>(Qh, Kh, Vth, Ab);

    // Output projection -> fp32 d_out
    out_gemm_f32<<<128, 256, 0, stream>>>(Ab, Wob, b_o, (float*)d_out, 1024, 1024, 16);
}
